// linear_retention_26989574488784
// MI455X (gfx1250) — matrix operand order repaired and checked by reading, not compiled, not run on hardware
//
#include <hip/hip_runtime.h>
#include <hip/hip_bf16.h>

typedef __attribute__((ext_vector_type(16))) _Float16 v16h;
typedef __attribute__((ext_vector_type(8)))  _Float16 v8h;
typedef __attribute__((ext_vector_type(4)))  _Float16 v4h;
typedef __attribute__((ext_vector_type(8)))  float    v8f;
typedef __attribute__((ext_vector_type(4)))  unsigned int v4u;

#define M_DIM 16384   // H*W = 128*128
#define E_DIM 1024
#define NHEADS 16
#define HDIM 64

// ---------------------------------------------------------------------------
// CDNA5 async global->LDS copy (ASYNCcnt path) via inline asm.
// vdst = per-lane LDS byte address (low 32 bits of the flat pointer, per the
// LDS aperture rule addr[31:0]); vaddr = 64-bit global address.
// ---------------------------------------------------------------------------
__device__ __forceinline__ void copy16_to_lds(const _Float16* g, _Float16* l) {
    unsigned lds = (unsigned)(unsigned long long)(const void*)l;
    unsigned long long ga = (unsigned long long)(const void*)g;
    asm volatile("global_load_async_to_lds_b128 %0, %1, off"
                 :: "v"(lds), "v"(ga)
                 : "memory");
}

__device__ __forceinline__ void wait_async_all() {
    asm volatile("s_wait_asynccnt 0x0" ::: "memory");
}

// ---------------------------------------------------------------------------
// fp32 -> fp16 conversion (vectorized grid-stride)
// ---------------------------------------------------------------------------
__global__ void cvt_f32_f16(const float* __restrict__ s, _Float16* __restrict__ d, int n) {
    int i = (blockIdx.x * blockDim.x + threadIdx.x) * 4;
    int stride = gridDim.x * blockDim.x * 4;
    for (; i + 3 < n; i += stride) {
        float4 f = *(const float4*)(s + i);
        v4h h;
        h[0] = (_Float16)f.x; h[1] = (_Float16)f.y;
        h[2] = (_Float16)f.z; h[3] = (_Float16)f.w;
        *(v4h*)(d + i) = h;
    }
}

// ---------------------------------------------------------------------------
// Tiled f16 GEMM:  out[16384 x 1024] = A[16384 x 1024] * W[1024 x 1024]^T
//                  + bias, optional ReLU.
// Block tile 128x128, BK=32, 256 threads (8 waves), wave tile 16x128.
// Double-buffered LDS: async loads for tile i+1 overlap WMMA on tile i.
// ---------------------------------------------------------------------------
#define GLD 40   // 32 + 8 pad (f16 units)

__global__ __launch_bounds__(256)
void gemm_bias_f16(const _Float16* __restrict__ A,
                   const _Float16* __restrict__ W,
                   const float*    __restrict__ bias,
                   _Float16* __restrict__ outH,
                   float*    __restrict__ outF,
                   int relu)
{
    __shared__ _Float16 As[2][128 * GLD];
    __shared__ _Float16 Bs[2][128 * GLD];
    const int t    = threadIdx.x;
    const int wave = t >> 5;
    const int lane = t & 31;
    const int half = lane >> 4;
    const int l16  = lane & 15;
    const int rowbase = blockIdx.x * 128;
    const int colbase = blockIdx.y * 128;

    v8f acc[8] = {};

    // prologue: issue async fill of buffer 0 (K-tile 0)
    #pragma unroll
    for (int i = 0; i < 2; ++i) {
        int e = t + i * 256;            // 0..511
        int r = e >> 2;                 // row 0..127
        int c = (e & 3) * 8;            // f16 col chunk
        copy16_to_lds(&A[(size_t)(rowbase + r) * E_DIM + c], &As[0][r * GLD + c]);
        copy16_to_lds(&W[(size_t)(colbase + r) * E_DIM + c], &Bs[0][r * GLD + c]);
    }

    int cur = 0;
    #pragma unroll 2
    for (int k0 = 0; k0 < E_DIM; k0 += 32, cur ^= 1) {
        wait_async_all();      // current buffer's data has landed in LDS
        __syncthreads();       // all waves done reading the buffer we refill

        if (k0 + 32 < E_DIM) {
            // async fill of the other buffer overlaps this tile's WMMAs
            #pragma unroll
            for (int i = 0; i < 2; ++i) {
                int e = t + i * 256;
                int r = e >> 2;
                int c = (e & 3) * 8;
                copy16_to_lds(&A[(size_t)(rowbase + r) * E_DIM + k0 + 32 + c],
                              &As[cur ^ 1][r * GLD + c]);
                copy16_to_lds(&W[(size_t)(colbase + r) * E_DIM + k0 + 32 + c],
                              &Bs[cur ^ 1][r * GLD + c]);
            }
        }
        if (k0 + 64 <= E_DIM) {
            // L2 prefetch of the tile after next
            #pragma unroll
            for (int i = 0; i < 2; ++i) {
                int e = t + i * 256;
                int r = e >> 2;
                int c = (e & 3) * 8;
                __builtin_prefetch(&A[(size_t)(rowbase + r) * E_DIM + k0 + 64 + c], 0, 3);
                __builtin_prefetch(&W[(size_t)(colbase + r) * E_DIM + k0 + 64 + c], 0, 3);
            }
        }

        union F { v16h v; v8h h[2]; };
        F af;
        const int am = wave * 16 + l16;
        af.h[0] = *(const v8h*)&As[cur][am * GLD + 8 * half];
        af.h[1] = *(const v8h*)&As[cur][am * GLD + 16 + 8 * half];

        F bf[8];
        #pragma unroll
        for (int j = 0; j < 8; ++j) {
            const int bn = j * 16 + l16;
            bf[j].h[0] = *(const v8h*)&Bs[cur][bn * GLD + 8 * half];
            bf[j].h[1] = *(const v8h*)&Bs[cur][bn * GLD + 16 + 8 * half];
        }
        #pragma unroll
        for (int j = 0; j < 8; ++j)
            acc[j] = __builtin_amdgcn_wmma_f32_16x16x32_f16(
                false, af.v, false, bf[j].v, (short)0, acc[j], false, false);
    }

    // epilogue: bias (+ReLU), store f16 or f32
    #pragma unroll
    for (int j = 0; j < 8; ++j) {
        const int col = colbase + j * 16 + l16;
        const float bv = bias[col];
        #pragma unroll
        for (int r = 0; r < 8; ++r) {
            const int row = rowbase + wave * 16 + r + 8 * half;
            float val = acc[j][r] + bv;
            if (relu) val = fmaxf(val, 0.0f);
            if (outH) outH[(size_t)row * E_DIM + col] = (_Float16)val;
            else      outF[(size_t)row * E_DIM + col] = val;
        }
    }
}

// ---------------------------------------------------------------------------
// Retention state:  kv[d][e] = sum_w k[w][d] * v[w][e]   (per block = one
// (line, head) pair).  Slabs are 128x64 f16, staged transposed in LDS so
// both the A (=k^T) and B (=v) fragments are contiguous reads.  The 64x64
// result is stored TRANSPOSED (kvT[e][d]) so the apply kernel needs no
// transpose for its B matrix.
// ---------------------------------------------------------------------------
#define TLD 136  // 128 + 8 pad

__global__ __launch_bounds__(128)
void ret_kv_f16(const _Float16* __restrict__ Kb,
                const _Float16* __restrict__ Vb,
                _Float16* __restrict__ kvT,
                long long blockStride,   // element offset per blockIdx.x
                long long rowStride)     // element stride along reduction axis
{
    __shared__ _Float16 kT[64 * TLD];
    __shared__ _Float16 vT[64 * TLD];
    const int t    = threadIdx.x;
    const int wave = t >> 5;
    const int lane = t & 31;
    const int half = lane >> 4;
    const int l16  = lane & 15;
    const long long base = (long long)blockIdx.x * blockStride + (long long)blockIdx.y * HDIM;

    // load 128x64 slabs, transpose into LDS [d][w]
    #pragma unroll
    for (int i = 0; i < 8; ++i) {
        int idx = t + i * 128;        // 0..1023
        int w  = idx >> 3;            // 0..127
        int dc = (idx & 7) * 8;       // d chunk
        v8h kk = *(const v8h*)&Kb[base + (long long)w * rowStride + dc];
        v8h vv = *(const v8h*)&Vb[base + (long long)w * rowStride + dc];
        #pragma unroll
        for (int j = 0; j < 8; ++j) {
            kT[(dc + j) * TLD + w] = kk[j];
            vT[(dc + j) * TLD + w] = vv[j];
        }
    }
    __syncthreads();

    v8f acc[4] = {};
    const int am = wave * 16 + l16;       // d row strip
    #pragma unroll
    for (int k0 = 0; k0 < 128; k0 += 32) {
        union F { v16h v; v8h h[2]; };
        F af;
        af.h[0] = *(const v8h*)&kT[am * TLD + k0 + 8 * half];
        af.h[1] = *(const v8h*)&kT[am * TLD + k0 + 16 + 8 * half];
        F bf[4];
        #pragma unroll
        for (int j = 0; j < 4; ++j) {
            const int bn = j * 16 + l16;
            bf[j].h[0] = *(const v8h*)&vT[bn * TLD + k0 + 8 * half];
            bf[j].h[1] = *(const v8h*)&vT[bn * TLD + k0 + 16 + 8 * half];
        }
        #pragma unroll
        for (int j = 0; j < 4; ++j)
            acc[j] = __builtin_amdgcn_wmma_f32_16x16x32_f16(
                false, af.v, false, bf[j].v, (short)0, acc[j], false, false);
    }

    _Float16* outp = kvT + ((size_t)blockIdx.x * NHEADS + blockIdx.y) * (HDIM * HDIM);
    #pragma unroll
    for (int j = 0; j < 4; ++j) {
        const int e = j * 16 + l16;
        #pragma unroll
        for (int r = 0; r < 8; ++r) {
            const int d = wave * 16 + r + 8 * half;
            outp[e * HDIM + d] = (_Float16)acc[j][r];   // transposed store
        }
    }
}

// ---------------------------------------------------------------------------
// Retention apply:  out[m][e] = sum_d A[m][d] * kv[d][e], kv given transposed.
// A slab 128x64 natural layout; B rows = kvT[e][d] contiguous.
// 128 threads = 4 waves, each wave does 2 row strips x 4 col tiles.
// ---------------------------------------------------------------------------
#define QLD 72   // 64 + 8 pad

__global__ __launch_bounds__(128)
void ret_apply_f16(const _Float16* __restrict__ Ab,
                   const _Float16* __restrict__ kvT,
                   _Float16* __restrict__ Ob,
                   long long aBlockStride, long long aRowStride,
                   long long oBlockStride, long long oRowStride)
{
    __shared__ _Float16 Aq[128 * QLD];
    __shared__ _Float16 Bk[64 * QLD];
    const int t    = threadIdx.x;
    const int wave = t >> 5;
    const int lane = t & 31;
    const int half = lane >> 4;
    const int l16  = lane & 15;
    const long long abase = (long long)blockIdx.x * aBlockStride + (long long)blockIdx.y * HDIM;
    const long long obase = (long long)blockIdx.x * oBlockStride + (long long)blockIdx.y * HDIM;
    const _Float16* kvp = kvT + ((size_t)blockIdx.x * NHEADS + blockIdx.y) * (HDIM * HDIM);

    #pragma unroll
    for (int i = 0; i < 8; ++i) {
        int idx = t + i * 128;        // 0..1023
        int m  = idx >> 3;
        int dc = (idx & 7) * 8;
        copy16_to_lds(&Ab[abase + (long long)m * aRowStride + dc], &Aq[m * QLD + dc]);
    }
    #pragma unroll
    for (int i = 0; i < 4; ++i) {
        int idx = t + i * 128;        // 0..511
        int r  = idx >> 3;            // 0..63
        int dc = (idx & 7) * 8;
        copy16_to_lds(&kvp[r * HDIM + dc], &Bk[r * QLD + dc]);
    }
    wait_async_all();
    __syncthreads();

    v8f acc[2][4] = {};
    #pragma unroll
    for (int k0 = 0; k0 < 64; k0 += 32) {
        #pragma unroll
        for (int s = 0; s < 2; ++s) {
            const int am = wave * 32 + s * 16 + l16;
            union F { v16h v; v8h h[2]; };
            F af;
            af.h[0] = *(const v8h*)&Aq[am * QLD + k0 + 8 * half];
            af.h[1] = *(const v8h*)&Aq[am * QLD + k0 + 16 + 8 * half];
            F bf[4];
            #pragma unroll
            for (int j = 0; j < 4; ++j) {
                const int bn = j * 16 + l16;
                bf[j].h[0] = *(const v8h*)&Bk[bn * QLD + k0 + 8 * half];
                bf[j].h[1] = *(const v8h*)&Bk[bn * QLD + k0 + 16 + 8 * half];
            }
            #pragma unroll
            for (int j = 0; j < 4; ++j)
                acc[s][j] = __builtin_amdgcn_wmma_f32_16x16x32_f16(
                    false, af.v, false, bf[j].v, (short)0, acc[s][j], false, false);
        }
    }

    #pragma unroll
    for (int s = 0; s < 2; ++s)
        #pragma unroll
        for (int j = 0; j < 4; ++j) {
            const int e = j * 16 + l16;
            #pragma unroll
            for (int r = 0; r < 8; ++r) {
                const int m = wave * 32 + s * 16 + r + 8 * half;
                Ob[obase + (long long)m * oRowStride + e] = (_Float16)acc[s][j][r];
            }
        }
}

// ---------------------------------------------------------------------------
// launch
// ---------------------------------------------------------------------------
extern "C" void kernel_launch(void* const* d_in, const int* in_sizes, int n_in,
                              void* d_out, int out_size, void* d_ws, size_t ws_size,
                              hipStream_t stream)
{
    const float* x  = (const float*)d_in[0];
    const float* Wq = (const float*)d_in[1];
    const float* bq = (const float*)d_in[2];
    const float* Wk = (const float*)d_in[3];
    const float* bk = (const float*)d_in[4];
    const float* Wv = (const float*)d_in[5];
    const float* bv = (const float*)d_in[6];
    const float* Wo = (const float*)d_in[7];
    const float* bo = (const float*)d_in[8];
    float* out = (float*)d_out;

    const size_t ME = (size_t)M_DIM * E_DIM;   // 16.78M
    const size_t EE = (size_t)E_DIM * E_DIM;   // 1.05M
    const size_t KV = (size_t)128 * NHEADS * HDIM * HDIM; // 8.39M

    char* p = (char*)d_ws;
    _Float16* x_h  = (_Float16*)p; p += ME * 2;
    _Float16* wq_h = (_Float16*)p; p += EE * 2;
    _Float16* wk_h = (_Float16*)p; p += EE * 2;
    _Float16* wv_h = (_Float16*)p; p += EE * 2;
    _Float16* wo_h = (_Float16*)p; p += EE * 2;
    _Float16* q_h  = (_Float16*)p; p += ME * 2;
    _Float16* k_h  = (_Float16*)p; p += ME * 2;
    _Float16* v_h  = (_Float16*)p; p += ME * 2;
    _Float16* kvT  = (_Float16*)p; p += KV * 2;
    _Float16* o1   = (_Float16*)p; p += ME * 2;
    _Float16* o2   = q_h;   // q is dead after width-apply; reuse

    // fp32 -> fp16
    cvt_f32_f16<<<8192, 256, 0, stream>>>(x,  x_h,  (int)ME);
    cvt_f32_f16<<<1024, 256, 0, stream>>>(Wq, wq_h, (int)EE);
    cvt_f32_f16<<<1024, 256, 0, stream>>>(Wk, wk_h, (int)EE);
    cvt_f32_f16<<<1024, 256, 0, stream>>>(Wv, wv_h, (int)EE);
    cvt_f32_f16<<<1024, 256, 0, stream>>>(Wo, wo_h, (int)EE);

    // Q/K/V projections (+ReLU on q,k)
    dim3 gg(M_DIM / 128, E_DIM / 128);
    gemm_bias_f16<<<gg, 256, 0, stream>>>(x_h, wq_h, bq, q_h, nullptr, 1);
    gemm_bias_f16<<<gg, 256, 0, stream>>>(x_h, wk_h, bk, k_h, nullptr, 1);
    gemm_bias_f16<<<gg, 256, 0, stream>>>(x_h, wv_h, bv, v_h, nullptr, 0);

    dim3 gr(128, NHEADS);
    // width-axis retention: per (row h, head): reduce over w (row stride 1024)
    ret_kv_f16   <<<gr, 128, 0, stream>>>(k_h, v_h, kvT, 131072LL, 1024LL);
    ret_apply_f16<<<gr, 128, 0, stream>>>(q_h, kvT, o1, 131072LL, 1024LL, 131072LL, 1024LL);
    // height-axis retention: per (col w, head): reduce over h (row stride 131072)
    ret_kv_f16   <<<gr, 128, 0, stream>>>(k_h, v_h, kvT, 1024LL, 131072LL);
    ret_apply_f16<<<gr, 128, 0, stream>>>(o1, kvT, o2, 1024LL, 131072LL, 1024LL, 131072LL);

    // output projection, fp32 result
    gemm_bias_f16<<<gg, 256, 0, stream>>>(o2, wo_h, bo, nullptr, out, 0);
}
